// Attention_19035295056566
// MI455X (gfx1250) — matrix operand order repaired and checked by reading, not compiled, not run on hardware
//
#include <hip/hip_runtime.h>
#include <hip/hip_bf16.h>
#include <stdint.h>

typedef __attribute__((ext_vector_type(16))) __bf16 v16bf;
typedef __attribute__((ext_vector_type(8)))  float  v8f;

union Frag {
    v16bf v;
    uint4 u[2];
};

#define XS 1552            // LDS bytes per A-tile row: 768 bf16 + 8 pad
#define QS 2312            // bf16 elems per LDS qkv row: 2304 + 8 pad

__device__ __forceinline__ float bflo(unsigned u) { return __uint_as_float(u << 16); }
__device__ __forceinline__ float bfhi(unsigned u) { return __uint_as_float(u & 0xffff0000u); }

// ---------------- weight conversion (one-time, tiny) ----------------
// in: (R x Cc) fp32 row-major  ->  out: (Cc x R) bf16 row-major (K-contiguous per column)
__global__ void k_cvt_T(const float* __restrict__ in, __bf16* __restrict__ out, int R, int Cc) {
    int i = blockIdx.x * blockDim.x + threadIdx.x;
    int total = R * Cc;
    if (i < total) {
        int r = i / Cc;
        int c = i - r * Cc;
        out[(long)c * R + r] = (__bf16)in[i];
    }
}

// ---------------- fused QKV GEMM + per-token head attention ----------------
// grid: B * (N/32) = 1024 blocks, 512 threads (16 waves)
// wave w owns qkv cols [w*144, w*144+144) (9 n-tiles) x all 32 rows (2 m-tiles),
// so every B fragment is reused for two WMMAs.

__global__ __launch_bounds__(512) void k_qkv_attn(
    const float* __restrict__ x,        // (B,N,768) fp32 (converted to bf16 while staging)
    const __bf16* __restrict__ wqkvT,   // (2304,768) bf16, K-contiguous
    __bf16* __restrict__ ctx)           // (B,H,N,D) bf16 == shuffled proj input
{
    extern __shared__ uint4 smem4[];
    char*   smem   = (char*)smem4;
    char*   sh_x   = smem;                                   // 32*1552   = 49664 B
    __bf16* sh_qkv = (__bf16*)(smem + 32 * XS);              // 32*2312*2 = 147968 B
    float*  sh_log = (float*)(smem + 32 * XS + 32 * QS * 2); // 32*36*4   = 4608 B

    const int b     = blockIdx.x >> 7;
    const int n0    = (blockIdx.x & 127) * 32;
    const int tid   = threadIdx.x;
    const int lane  = tid & 31;
    const int wave  = tid >> 5;
    const int col15 = lane & 15;
    const int hi    = (lane >> 4) & 1;

    // ---- stage X tile (32 x 768) into LDS, converting fp32 -> bf16 on the fly ----
    const float* xsrc = x + ((long)b * 4096 + n0) * 768;
    for (int i = tid; i < 32 * 192; i += 512) {     // 192 float4 per row
        int r = i / 192, c = i - r * 192;
        float4 f = *(const float4*)(xsrc + (long)r * 768 + c * 4);
        union { __bf16 bb[4]; uint2 u; } o;
        o.bb[0] = (__bf16)f.x; o.bb[1] = (__bf16)f.y;
        o.bb[2] = (__bf16)f.z; o.bb[3] = (__bf16)f.w;
        *(uint2*)(sh_x + r * XS + c * 8) = o.u;
    }
    __syncthreads();

    // ---- GEMM: qkv(32x2304) = X(32x768) @ Wqkv ----
    const int colbase = wave * 144;                 // 9 n-tiles per wave
    const char*   arow0 = sh_x + col15 * XS;        // rows 0..15
    const char*   arow1 = sh_x + (16 + col15) * XS; // rows 16..31
    const __bf16* bbase = wqkvT + (long)(colbase + col15) * 768 + hi * 8;

    v8f acc0[9] = {};
    v8f acc1[9] = {};
    for (int k0 = 0; k0 < 768; k0 += 32) {
        Frag a0, a1;
        a0.u[0] = *(const uint4*)(arow0 + 2 * (k0 + hi * 8));
        a0.u[1] = *(const uint4*)(arow0 + 2 * (k0 + hi * 8) + 32);
        a1.u[0] = *(const uint4*)(arow1 + 2 * (k0 + hi * 8));
        a1.u[1] = *(const uint4*)(arow1 + 2 * (k0 + hi * 8) + 32);
#pragma unroll
        for (int t = 0; t < 9; ++t) {
            Frag bm;
            const __bf16* bp = bbase + (long)t * 16 * 768 + k0;
            bm.u[0] = *(const uint4*)(bp);
            bm.u[1] = *(const uint4*)(bp + 16);
            acc0[t] = __builtin_amdgcn_wmma_f32_16x16x32_bf16(
                false, a0.v, false, bm.v, (short)0, acc0[t], false, false);
            acc1[t] = __builtin_amdgcn_wmma_f32_16x16x32_bf16(
                false, a1.v, false, bm.v, (short)0, acc1[t], false, false);
        }
    }

    // ---- spill accumulators to LDS qkv staging (bf16) ----
#pragma unroll
    for (int t = 0; t < 9; ++t) {
        int col = colbase + t * 16 + col15;
#pragma unroll
        for (int r = 0; r < 8; ++r) {
            int row = r + hi * 8;                  // C/D layout: VGPR r -> M = r + 8*hi
            sh_qkv[row * QS + col]        = (__bf16)acc0[t][r];
            sh_qkv[(16 + row) * QS + col] = (__bf16)acc1[t][r];
        }
    }
    __syncthreads();

    // ---- per-token 6x6 head attention (fp32) ----
    const float scale = 0.08838834764831845f;   // 128^-0.5
    {
        int tok = tid >> 4;
        int j   = tid & 15;
        const __bf16* qr = sh_qkv + tok * QS;
        for (int p = j; p < 36; p += 16) {
            int h = p / 6, g = p - (p / 6) * 6;
            const __bf16* q = qr + h * 128;
            const __bf16* k = qr + 768 + g * 128;
            float s = 0.f;
#pragma unroll
            for (int d = 0; d < 128; d += 2) {
                unsigned uq = *(const unsigned*)(q + d);
                unsigned uk = *(const unsigned*)(k + d);
                s += bflo(uq) * bflo(uk) + bfhi(uq) * bfhi(uk);
            }
            sh_log[tok * 36 + p] = s * scale;
        }
    }
    __syncthreads();
    {
        int tok = tid >> 4;
        int j   = tid & 15;
        if (j < 6) {
            float* lr = sh_log + tok * 36 + j * 6;
            float m = lr[0];
#pragma unroll
            for (int g = 1; g < 6; ++g) m = fmaxf(m, lr[g]);
            float e[6], sum = 0.f;
#pragma unroll
            for (int g = 0; g < 6; ++g) { e[g] = __expf(lr[g] - m); sum += e[g]; }
            float inv = 1.f / sum;
#pragma unroll
            for (int g = 0; g < 6; ++g) lr[g] = e[g] * inv;
        }
    }
    __syncthreads();

    // ---- context = attn @ v, written in (B,H,N,D) order (== shuffled layout) ----
    for (int i = tid; i < 32 * 384; i += 512) {     // 384 d-pairs per token
        int tok   = i / 384;
        int cpair = i - tok * 384;
        int h  = cpair >> 6;                        // 64 pairs per head
        int dp = (cpair & 63) * 2;
        const float*  w = sh_log + tok * 36 + h * 6;
        const __bf16* v = sh_qkv + tok * QS + 1536;
        float s0 = 0.f, s1 = 0.f;
#pragma unroll
        for (int g = 0; g < 6; ++g) {
            unsigned uv = *(const unsigned*)(v + g * 128 + dp);
            s0 += w[g] * bflo(uv);
            s1 += w[g] * bfhi(uv);
        }
        union { __bf16 bb[2]; unsigned u; } o;
        o.bb[0] = (__bf16)s0; o.bb[1] = (__bf16)s1;
        *(unsigned*)(ctx + (((long)b * 6 + h) * 4096 + (n0 + tok)) * 128 + dp) = o.u;
    }
}

// ---------------- output projection: Y = ctx_shuffled @ Wproj + bias ----------------
// grid: B * (N/32) = 1024 blocks, 512 threads (16 waves)
// wave w owns cols [w*48, w*48+48) (3 n-tiles) x 32 rows (2 m-tiles): B frag reused 2x.

__global__ __launch_bounds__(512) void k_proj(
    const __bf16* __restrict__ ctx,      // (B, 4096, 768) bf16 (shuffled rows)
    const __bf16* __restrict__ wprojT,   // (768, 768) bf16, K-contiguous
    const float* __restrict__ bias,
    float* __restrict__ out)             // (B, 4096, 768) fp32
{
    extern __shared__ uint4 smem4[];
    char* sh_a = (char*)smem4;           // 32 * XS

    const int b     = blockIdx.x >> 7;
    const int m0    = (blockIdx.x & 127) * 32;
    const int tid   = threadIdx.x;
    const int lane  = tid & 31;
    const int wave  = tid >> 5;
    const int col15 = lane & 15;
    const int hi    = (lane >> 4) & 1;

    const __bf16* asrc = ctx + ((long)b * 4096 + m0) * 768;
    for (int i = tid; i < 32 * 48; i += 512) {
        int r = i / 48, c = i - r * 48;
        *(uint4*)(sh_a + r * XS + c * 16) = *(const uint4*)(asrc + (long)r * 768 + c * 8);
    }
    __syncthreads();

    const int colbase = wave * 48;                  // 3 n-tiles per wave
    const char*   arow0 = sh_a + col15 * XS;
    const char*   arow1 = sh_a + (16 + col15) * XS;
    const __bf16* bbase = wprojT + (long)(colbase + col15) * 768 + hi * 8;

    v8f acc0[3] = {};
    v8f acc1[3] = {};
    for (int k0 = 0; k0 < 768; k0 += 32) {
        Frag a0, a1;
        a0.u[0] = *(const uint4*)(arow0 + 2 * (k0 + hi * 8));
        a0.u[1] = *(const uint4*)(arow0 + 2 * (k0 + hi * 8) + 32);
        a1.u[0] = *(const uint4*)(arow1 + 2 * (k0 + hi * 8));
        a1.u[1] = *(const uint4*)(arow1 + 2 * (k0 + hi * 8) + 32);
#pragma unroll
        for (int t = 0; t < 3; ++t) {
            Frag bm;
            const __bf16* bp = bbase + (long)t * 16 * 768 + k0;
            bm.u[0] = *(const uint4*)(bp);
            bm.u[1] = *(const uint4*)(bp + 16);
            acc0[t] = __builtin_amdgcn_wmma_f32_16x16x32_bf16(
                false, a0.v, false, bm.v, (short)0, acc0[t], false, false);
            acc1[t] = __builtin_amdgcn_wmma_f32_16x16x32_bf16(
                false, a1.v, false, bm.v, (short)0, acc1[t], false, false);
        }
    }

#pragma unroll
    for (int t = 0; t < 3; ++t) {
        int col = colbase + t * 16 + col15;
        float bv = bias[col];
#pragma unroll
        for (int r = 0; r < 8; ++r) {
            int row = r + hi * 8;
            out[((long)b * 4096 + m0 + row) * 768 + col]      = acc0[t][r] + bv;
            out[((long)b * 4096 + m0 + 16 + row) * 768 + col] = acc1[t][r] + bv;
        }
    }
}

// ---------------- host launcher ----------------

extern "C" void kernel_launch(void* const* d_in, const int* in_sizes, int n_in,
                              void* d_out, int out_size, void* d_ws, size_t ws_size,
                              hipStream_t stream) {
    const float* x     = (const float*)d_in[0];   // (8, 4096, 768)
    const float* wqkv  = (const float*)d_in[1];   // (768, 2304)
    const float* wproj = (const float*)d_in[2];   // (768, 768)
    const float* bproj = (const float*)d_in[3];   // (768,)
    float* out = (float*)d_out;

    char* ws = (char*)d_ws;
    __bf16* wqkvT  = (__bf16*)(ws);                 // 2304*768*2 = 3,538,944 B
    __bf16* wprojT = (__bf16*)(ws + 3538944);       //  768*768*2 = 1,179,648 B
    __bf16* ctx    = (__bf16*)(ws + 4718592);       // 25,165,824 elems (bf16)
    // total workspace need: ~55.1 MB

    {   // w_qkv (768x2304) -> transposed bf16 (2304x768)
        int tot = 768 * 2304;
        k_cvt_T<<<(tot + 255) / 256, 256, 0, stream>>>(wqkv, wqkvT, 768, 2304);
    }
    {   // w_proj (768x768) -> transposed bf16
        int tot = 768 * 768;
        k_cvt_T<<<(tot + 255) / 256, 256, 0, stream>>>(wproj, wprojT, 768, 768);
    }

    size_t ldsB = (size_t)32 * XS + (size_t)32 * QS * 2 + (size_t)32 * 36 * 4; // ~202 KB
    k_qkv_attn<<<1024, 512, ldsB, stream>>>(x, wqkvT, ctx);

    size_t ldsC = (size_t)32 * XS;                                             // ~48.5 KB
    k_proj<<<1024, 512, ldsC, stream>>>(ctx, wprojT, bproj, out);
}
